// RNNModule_85409719648806
// MI455X (gfx1250) — compile-verified, matrix operand order checked by reading, not run on hardware
//
#include <hip/hip_runtime.h>
#include <math.h>
#include <stdint.h>

// Problem constants (match reference)
#define B_ 128
#define T_ 512
#define I_ 128
#define H_ 256

// CDNA5 WMMA operand types
typedef __attribute__((ext_vector_type(16))) __bf16 v16bf;
typedef __attribute__((ext_vector_type(8)))  float  v8f;

// A/B fragment: 16 bf16 = 8 VGPRs. Per ISA 7.12.2 (16-bit A 16x32 and the
// symmetric B 32x16 layout): lane L holds row r=L&15 of the matrix, with
// K-base kb = (L>>4)*8; elements 0..7 = K[kb..kb+8), elements 8..15 =
// K[kb+16..kb+24). Both runs are 8 contiguous elements -> 16B aligned loads.
union Frag {
  v16bf    v;
  uint16_t h[16];
  uint4    q[2];
};

// gfx1250 has native BF16 conversion; let clang lower the fptrunc to the
// hardware v_cvt path instead of a 3-op software RNE sequence.
__device__ __forceinline__ uint16_t f32_to_bf16(float f) {
  return __builtin_bit_cast(uint16_t, static_cast<__bf16>(f));
}

// tanh via the hardware transcendental: tanh(x) = (e^2x - 1) / (e^2x + 1).
// Clamp keeps e^2x finite; at |x|=15 tanh is 1.0 to f32 precision anyway.
__device__ __forceinline__ float fast_tanh(float x) {
  x = fminf(fmaxf(x, -15.0f), 15.0f);
  const float e = __expf(2.0f * x);
  return (e - 1.0f) / (e + 1.0f);
}

// ---------------------------------------------------------------------------
// Phase A: xp[t][b][h] = sum_i x[b][t][i] * W_ih[h][i] + b_ih[h] + b_hh[h]
// One wave per 16x16 output tile; K=128 -> 4 bf16 WMMAs.
// Output layout [T][B][H] so the scan kernel streams contiguous slabs per t.
// ---------------------------------------------------------------------------
__global__ __launch_bounds__(256) void xproj_kernel(
    const float* __restrict__ x, const float* __restrict__ W_ih,
    const float* __restrict__ b_ih, const float* __restrict__ b_hh,
    float* __restrict__ xp)
{
  const int lane  = threadIdx.x & 31;
  const int wave  = threadIdx.x >> 5;
  const int gw    = blockIdx.x * 8 + wave;    // 65536 tiles total
  const int ntile = gw & 15;                  // 16 tiles across H=256
  const int mtile = gw >> 4;                  // 4096 tiles across B*T
  const int r     = lane & 15;
  const int kb    = (lane >> 4) << 3;         // 0 or 8

  const float* arow = x    + (size_t)(mtile * 16 + r) * I_;  // rows of x (b-major flat)
  const float* brow = W_ih + (size_t)(ntile * 16 + r) * I_;  // rows of W_ih = B operand

  v8f acc = {};
  #pragma unroll
  for (int c = 0; c < 4; ++c) {
    const int k0 = 32 * c + kb;
    const int k1 = k0 + 16;
    Frag a, b;
    #pragma unroll
    for (int i = 0; i < 8; ++i) {
      a.h[i]     = f32_to_bf16(arow[k0 + i]);
      a.h[8 + i] = f32_to_bf16(arow[k1 + i]);
      b.h[i]     = f32_to_bf16(brow[k0 + i]);
      b.h[8 + i] = f32_to_bf16(brow[k1 + i]);
    }
    acc = __builtin_amdgcn_wmma_f32_16x16x32_bf16(
        false, a.v, false, b.v, (short)0, acc, false, false);
  }

  // C/D layout: lane -> n = lane&15, VGPR v -> m = v + (lane>>4)*8
  const int   n    = ntile * 16 + (lane & 15);
  const float bias = b_ih[n] + b_hh[n];
  const int   mOff = (lane >> 4) << 3;
  #pragma unroll
  for (int v = 0; v < 8; ++v) {
    const int mrow = mtile * 16 + v + mOff;  // flat index into B*T (b-major)
    const int bb   = mrow >> 9;              // / T
    const int tt   = mrow & (T_ - 1);        // % T
    xp[((size_t)tt * B_ + bb) * H_ + n] = acc[v] + bias;
  }
}

// ---------------------------------------------------------------------------
// Phase B+C: sequential scan. 8 blocks x 16 batch rows; 16 waves/block, wave
// nt owns output columns [nt*16, nt*16+16). W_hh bf16 fragments pinned in
// VGPRs; h double-buffered in LDS as bf16 (ONE barrier per step); 8 chained
// WMMAs + fast tanh per step; next xp slab prefetched into cache.
// ---------------------------------------------------------------------------
__global__ __launch_bounds__(512) void rnn_scan_kernel(
    const float* __restrict__ xp, const float* __restrict__ W_hh,
    const float* __restrict__ W_fc, const float* __restrict__ b_fc,
    float* __restrict__ out)
{
  __shared__ uint16_t hbuf[2][16][H_];  // bf16 hidden state, double-buffered, 16KB
  __shared__ float    hfin[16][H_];     // f32 hidden at t=T-1 for the head, 16KB

  const int lane = threadIdx.x & 31;
  const int nt   = threadIdx.x >> 5;  // wave id == N tile
  const int r    = lane & 15;
  const int kb   = (lane >> 4) << 3;
  const int b0   = blockIdx.x * 16;

  // Pin W_hh B-fragments in registers: row = nt*16 + r of W_hh (row-major
  // [N][K] is exactly the B-operand layout), all K=256 -> 8 fragments.
  Frag wB[8];
  {
    const float* wrow = W_hh + (size_t)(nt * 16 + r) * H_;
    #pragma unroll
    for (int c = 0; c < 8; ++c) {
      const int k0 = 32 * c + kb, k1 = k0 + 16;
      #pragma unroll
      for (int i = 0; i < 8; ++i) {
        wB[c].h[i]     = f32_to_bf16(wrow[k0 + i]);
        wB[c].h[8 + i] = f32_to_bf16(wrow[k1 + i]);
      }
    }
  }

  // h0 = 0 (buffer 0)
  for (int idx = threadIdx.x; idx < 16 * H_; idx += blockDim.x)
    ((uint16_t*)hbuf[0])[idx] = 0;
  __syncthreads();

  const int n    = nt * 16 + (lane & 15);
  const int mOff = (lane >> 4) << 3;

  int cur = 0;
  for (int t = 0; t < T_; ++t) {
    // Prefetch next step's xp slab (cache hint; no counter, no VGPR return)
    if (t + 1 < T_) {
      const float* xn = xp + ((size_t)(t + 1) * B_ + b0) * H_;
      __builtin_prefetch(xn + (size_t)mOff * H_ + n, 0, 3);
      __builtin_prefetch(xn + (size_t)(mOff + 4) * H_ + n, 0, 3);
    }

    // Gather A fragments of h from LDS buffer `cur` (16B-aligned ds loads).
    // Nobody writes `cur` this step (writes go to `cur^1`), so no pre-barrier.
    Frag a[8];
    #pragma unroll
    for (int c = 0; c < 8; ++c) {
      const int k0 = 32 * c + kb, k1 = k0 + 16;
      a[c].q[0] = *(const uint4*)&hbuf[cur][r][k0];
      a[c].q[1] = *(const uint4*)&hbuf[cur][r][k1];
    }

    v8f acc = {};
    #pragma unroll
    for (int c = 0; c < 8; ++c)
      acc = __builtin_amdgcn_wmma_f32_16x16x32_bf16(
          false, a[c].v, false, wB[c].v, (short)0, acc, false, false);

    const float* xpt = xp + ((size_t)t * B_ + b0) * H_;  // contiguous 128KB slab
    #pragma unroll
    for (int v = 0; v < 8; ++v) {
      const int m = v + mOff;
      const float z = fast_tanh(acc[v] + xpt[(size_t)m * H_ + n]);
      hbuf[cur ^ 1][m][n] = f32_to_bf16(z);
      if (t == T_ - 1) hfin[m][n] = z;
    }
    __syncthreads();  // h_new (buffer cur^1) visible before next step's reads
    cur ^= 1;
  }

  // Head: logits[b] = h_last . W_fc + b_fc ; sigmoid. O=1, so 16 lanes suffice.
  if (threadIdx.x < 16) {
    float acc = b_fc[0];
    #pragma unroll 4
    for (int k = 0; k < H_; ++k)
      acc += hfin[threadIdx.x][k] * W_fc[k];
    out[b0 + threadIdx.x] = 1.0f / (1.0f + __expf(-acc));
  }
}

// ---------------------------------------------------------------------------
extern "C" void kernel_launch(void* const* d_in, const int* in_sizes, int n_in,
                              void* d_out, int out_size, void* d_ws, size_t ws_size,
                              hipStream_t stream) {
  const float* x    = (const float*)d_in[0];  // [B,T,I]
  const float* W_ih = (const float*)d_in[1];  // [H,I]
  const float* W_hh = (const float*)d_in[2];  // [H,H]
  const float* b_ih = (const float*)d_in[3];  // [H]
  const float* b_hh = (const float*)d_in[4];  // [H]
  const float* W_fc = (const float*)d_in[5];  // [O,H] = [1,256]
  const float* b_fc = (const float*)d_in[6];  // [O]

  float* xp  = (float*)d_ws;                  // [T][B][H] f32 = 64 MB scratch
  float* out = (float*)d_out;                 // [B] = 128 f32

  // Phase A: 65536 16x16 tiles, 8 waves per 256-thread block
  xproj_kernel<<<(B_ * T_ / 16) * (H_ / 16) / 8, 256, 0, stream>>>(
      x, W_ih, b_ih, b_hh, xp);

  // Phase B+C: persistent scan, 8 blocks x 512 threads (16 waves)
  rnn_scan_kernel<<<B_ / 16, 512, 0, stream>>>(xp, W_hh, W_fc, b_fc, out);
}
